// GATv2Net_80479097193054
// MI455X (gfx1250) — matrix operand order repaired and checked by reading, not compiled, not run on hardware
//
#include <hip/hip_runtime.h>

#define HEADS 4
#define CH 32
#define FDIM 128
#define LAYERS 3
#define NEG_SLOPE 0.2f
#define EPSV 1e-16f

#define AS1 __attribute__((address_space(1)))
#define AS3 __attribute__((address_space(3)))

typedef __attribute__((ext_vector_type(16))) __bf16 v16bf;
typedef __attribute__((ext_vector_type(8)))  float  v8f;
typedef __attribute__((ext_vector_type(4)))  int    int4v;
typedef __attribute__((ext_vector_type(8)))  unsigned short ushort8;
typedef __attribute__((ext_vector_type(16))) unsigned short ushort16;

union BFrag { ushort16 u; v16bf b; };

static __device__ __forceinline__ unsigned short f2bf(float f) {
    unsigned u = __float_as_uint(f);
    unsigned r = (u + 0x7FFFu + ((u >> 16) & 1u)) >> 16;   // RNE
    return (unsigned short)r;
}
static __device__ __forceinline__ unsigned f2ord(float f) {
    unsigned u = __float_as_uint(f);
    return (u & 0x80000000u) ? ~u : (u | 0x80000000u);
}
static __device__ __forceinline__ float ord2f(unsigned u) {
    return __uint_as_float((u & 0x80000000u) ? (u & 0x7FFFFFFFu) : ~u);
}

// ---------------- elementwise f32 -> bf16 ----------------
__global__ void k_f32_to_bf16(const float* __restrict__ src,
                              unsigned short* __restrict__ dst, int n) {
    int i = blockIdx.x * blockDim.x + threadIdx.x;
    if (i < n) dst[i] = f2bf(src[i]);
}

// ---------------- pack W[128][128] (row-major K,N) into WMMA B fragments ----
// Fragment t = nt*128 + kc*32 + lane holds 16 bf16 halves:
//   lanes 0-15 : n = nt*16+lane,    k = kc*32 + 0..7, 16..23
//   lanes 16-31: n = nt*16+lane-16, k = kc*32 + 8..15, 24..31
__global__ void k_pack_w(const float* __restrict__ W,
                         unsigned short* __restrict__ P) {
    int t = blockIdx.x * blockDim.x + threadIdx.x;
    if (t >= 8 * 4 * 32) return;
    int lane = t & 31;
    int kc   = (t >> 5) & 3;
    int nt   = t >> 7;
    int n    = nt * 16 + (lane & 15);
    int kbase = kc * 32 + ((lane >> 4) << 3);
    unsigned short* dst = P + t * 16;
#pragma unroll
    for (int i = 0; i < 16; i++) dst[i] = f2bf(W[(kbase + i + ((i >> 3) << 3)) * FDIM + n]);
}

// ---------------- GEMM: D[rows,128] = A_bf16[rows,128] x Wpacked + bias -----
// Packed weights (32 KB) staged once per block into LDS via the gfx1250
// async global->LDS path, then one wave computes a 16x128 strip:
// 8 N-tiles x 4 K-chunks = 32 WMMAs/wave with B fragments served from LDS.
__global__ void __launch_bounds__(256)
k_gemm_bf16(const unsigned short* __restrict__ A,
            const unsigned short* __restrict__ P,
            const float* __restrict__ bias,
            float* __restrict__ D, int rows) {
    __shared__ unsigned short lW[8 * 4 * 32 * 16];   // 16384 ushorts = 32 KB

    int tid  = threadIdx.x;
    // Cooperative stage: 2048 chunks of 16 B; 8 chunks per thread.
#if __has_builtin(__builtin_amdgcn_global_load_async_to_lds_b128) && \
    __has_builtin(__builtin_amdgcn_s_wait_asynccnt)
#pragma unroll
    for (int i = 0; i < 8; i++) {
        int c = tid + i * 256;
        __builtin_amdgcn_global_load_async_to_lds_b128(
            (AS1 int4v*)(P + (size_t)c * 8),
            (AS3 int4v*)&lW[(size_t)c * 8], 0, 0);
    }
    __builtin_amdgcn_s_wait_asynccnt(0);
#else
#pragma unroll
    for (int i = 0; i < 8; i++) {
        int c = tid + i * 256;
        *(ushort8*)&lW[(size_t)c * 8] = *(const ushort8*)(P + (size_t)c * 8);
    }
#endif
    __syncthreads();

    int wave = tid >> 5;
    int lane = tid & 31;
    int rowTile = blockIdx.x * 8 + wave;
    int row0 = rowTile * 16;
    if (row0 >= rows) return;                 // wave-uniform: EXEC stays all-1s
    int laneLo = lane & 15;
    int laneHi = lane >> 4;

    // A-fragment layout (16-bit A 16x32, ISA 7.12.2):
    // lanes 0-15 : halves 0-7 = K0+0..7,  halves 8-15 = K0+16..23
    // lanes 16-31: halves 0-7 = K0+8..15, halves 8-15 = K0+24..31
    const unsigned short* arow = A + (size_t)(row0 + laneLo) * FDIM + laneHi * 8;
    BFrag afrag[4];
#pragma unroll
    for (int kc = 0; kc < 4; kc++) {
        ushort8* ph = (ushort8*)&afrag[kc];
        ph[0] = *(const ushort8*)(arow + kc * 32);
        ph[1] = *(const ushort8*)(arow + kc * 32 + 16);
    }

#pragma unroll
    for (int nt = 0; nt < 8; nt++) {
        float bv = bias[nt * 16 + laneLo];
        v8f acc = {bv, bv, bv, bv, bv, bv, bv, bv};
#pragma unroll
        for (int kc = 0; kc < 4; kc++) {
            BFrag bf;
            bf.u = *(const ushort16*)&lW[((size_t)(nt * 4 + kc) * 32 + lane) * 16];
            acc = __builtin_amdgcn_wmma_f32_16x16x32_bf16(
                false, afrag[kc].b, false, bf.b, (short)0, acc, false, false);
        }
        // C/D layout: VGPR r -> row (row0 + laneHi*8 + r), col nt*16 + laneLo
        int col = nt * 16 + laneLo;
        int rbase = row0 + laneHi * 8;
#pragma unroll
        for (int r = 0; r < 8; r++)
            D[(size_t)(rbase + r) * FDIM + col] = acc[r];
    }
}

// ---------------- edge pass 1: score + segment max (ordered-uint atomicMax) -
__global__ void k_edge_score(const float* __restrict__ xl,
                             const float* __restrict__ xr,
                             const float* __restrict__ att,
                             const int* __restrict__ ei, int E, int Etot,
                             float* __restrict__ score,
                             unsigned* __restrict__ smax) {
    int t = blockIdx.x * blockDim.x + threadIdx.x;
    if (t >= Etot * HEADS) return;
    int e = t >> 2;
    int h = t & 3;
    int s, d;
    if (e < E) { s = ei[e]; d = ei[E + e]; } else { s = e - E; d = s; }
    const float4* pj = (const float4*)(xl + (size_t)s * FDIM + h * CH);
    const float4* pi = (const float4*)(xr + (size_t)d * FDIM + h * CH);
    const float4* pa = (const float4*)(att + h * CH);
    float acc = 0.f;
#pragma unroll
    for (int q = 0; q < 8; q++) {
        float4 a = pa[q], vj = pj[q], vi = pi[q];
        float x0 = vi.x + vj.x; x0 = x0 > 0.f ? x0 : NEG_SLOPE * x0;
        float x1 = vi.y + vj.y; x1 = x1 > 0.f ? x1 : NEG_SLOPE * x1;
        float x2 = vi.z + vj.z; x2 = x2 > 0.f ? x2 : NEG_SLOPE * x2;
        float x3 = vi.w + vj.w; x3 = x3 > 0.f ? x3 : NEG_SLOPE * x3;
        acc += a.x * x0 + a.y * x1 + a.z * x2 + a.w * x3;
    }
    score[t] = acc;
    atomicMax(&smax[d * HEADS + h], f2ord(acc));
}

// ---------------- edge pass 2: exp + segment sum ----------------------------
__global__ void k_edge_exp(const int* __restrict__ ei, int E, int Etot,
                           const unsigned* __restrict__ smax,
                           float* __restrict__ score /* in: score, out: expv */,
                           float* __restrict__ denom) {
    int t = blockIdx.x * blockDim.x + threadIdx.x;
    if (t >= Etot * HEADS) return;
    int e = t >> 2;
    int h = t & 3;
    int d = (e < E) ? ei[E + e] : (e - E);
    float m = ord2f(smax[d * HEADS + h]);
    float v = __expf(score[t] - m);
    score[t] = v;
    atomicAdd(&denom[d * HEADS + h], v);
}

// ---------------- edge pass 3: alpha-weighted scatter-add -------------------
__global__ void k_edge_aggr(const int* __restrict__ ei, int E, int Etot,
                            const float* __restrict__ xl,
                            const float* __restrict__ expv,
                            const float* __restrict__ denom,
                            float* __restrict__ out) {
    int t = blockIdx.x * blockDim.x + threadIdx.x;
    if (t >= Etot * HEADS) return;
    int e = t >> 2;
    int h = t & 3;
    int s, d;
    if (e < E) { s = ei[e]; d = ei[E + e]; } else { s = e - E; d = s; }
    float alpha = expv[t] / (denom[d * HEADS + h] + EPSV);
    const float4* pj = (const float4*)(xl + (size_t)s * FDIM + h * CH);
    float* po = out + (size_t)d * FDIM + h * CH;
#pragma unroll
    for (int q = 0; q < 8; q++) {
        float4 vj = pj[q];
        atomicAdd(po + q * 4 + 0, vj.x * alpha);
        atomicAdd(po + q * 4 + 1, vj.y * alpha);
        atomicAdd(po + q * 4 + 2, vj.z * alpha);
        atomicAdd(po + q * 4 + 3, vj.w * alpha);
    }
}

// ---------------- conv bias + relu ------------------------------------------
__global__ void k_bias_relu(const float* __restrict__ acc,
                            const float* __restrict__ bias,
                            float* __restrict__ out, int n) {
    int i = blockIdx.x * blockDim.x + threadIdx.x;
    if (i < n) {
        float v = acc[i] + bias[i & (FDIM - 1)];
        out[i] = v > 0.f ? v : 0.f;
    }
}

// ---------------- final linear head [N,128] @ [128,1] + bf ------------------
__global__ void k_head(const float* __restrict__ hbuf,
                       const float* __restrict__ Wf,
                       const float* __restrict__ bf,
                       float* __restrict__ out, int rows) {
    int wave = threadIdx.x >> 5;
    int lane = threadIdx.x & 31;
    int row = blockIdx.x * 8 + wave;
    if (row >= rows) return;
    const float* hr = hbuf + (size_t)row * FDIM;
    float v = 0.f;
#pragma unroll
    for (int q = 0; q < 4; q++) v += hr[lane + q * 32] * Wf[lane + q * 32];
#pragma unroll
    for (int off = 16; off > 0; off >>= 1) v += __shfl_xor(v, off, 32);
    if (lane == 0) out[row] = v + bf[0];
}

extern "C" void kernel_launch(void* const* d_in, const int* in_sizes, int n_in,
                              void* d_out, int out_size, void* d_ws, size_t ws_size,
                              hipStream_t stream) {
    const float* x    = (const float*)d_in[0];
    const int*   ei   = (const int*)d_in[1];
    // d_in[2] edge_weight: unused by GATv2 reference
    const float* Wl   = (const float*)d_in[3];
    const float* bl   = (const float*)d_in[4];
    const float* Wr   = (const float*)d_in[5];
    const float* br   = (const float*)d_in[6];
    const float* att  = (const float*)d_in[7];
    const float* bias = (const float*)d_in[8];
    const float* Wf   = (const float*)d_in[9];
    const float* bf   = (const float*)d_in[10];
    float* out = (float*)d_out;

    int Nn   = in_sizes[0] / FDIM;   // 50000
    int E    = in_sizes[1] / 2;      // 800000
    int Etot = E + Nn;               // self loops appended

    char* base = (char*)d_ws;
    size_t off = 0;
    auto alloc = [&](size_t bytes) -> char* {
        char* p = base + off;
        off += (bytes + 255) & ~(size_t)255;
        return p;
    };
    unsigned short* xbf  = (unsigned short*)alloc((size_t)Nn * FDIM * 2);
    unsigned short* wpk  = (unsigned short*)alloc((size_t)6 * 16384 * 2);
    float*    xl    = (float*)   alloc((size_t)Nn * FDIM * 4);
    float*    xr    = (float*)   alloc((size_t)Nn * FDIM * 4);
    float*    score = (float*)   alloc((size_t)Etot * HEADS * 4);
    unsigned* smax  = (unsigned*)alloc((size_t)Nn * HEADS * 4);
    float*    denom = (float*)   alloc((size_t)Nn * HEADS * 4);
    float*    accb  = (float*)   alloc((size_t)Nn * FDIM * 4);
    float*    hbuf  = (float*)   alloc((size_t)Nn * FDIM * 4);

    // pack all layer weights into WMMA B-fragment order (tiny, runs each call)
    for (int l = 0; l < LAYERS; l++) {
        k_pack_w<<<4, 256, 0, stream>>>(Wl + (size_t)l * FDIM * FDIM, wpk + (size_t)l * 16384);
        k_pack_w<<<4, 256, 0, stream>>>(Wr + (size_t)l * FDIM * FDIM, wpk + (size_t)(3 + l) * 16384);
    }

    int nElem = Nn * FDIM;
    int gElem = (nElem + 255) / 256;
    int gRows = ((Nn + 15) / 16 + 7) / 8;
    int gEdge = (Etot * HEADS + 255) / 256;

    const float* hin = x;
    for (int l = 0; l < LAYERS; l++) {
        k_f32_to_bf16<<<gElem, 256, 0, stream>>>(hin, xbf, nElem);
        k_gemm_bf16<<<gRows, 256, 0, stream>>>(xbf, wpk + (size_t)l * 16384,
                                               bl + l * FDIM, xl, Nn);
        k_gemm_bf16<<<gRows, 256, 0, stream>>>(xbf, wpk + (size_t)(3 + l) * 16384,
                                               br + l * FDIM, xr, Nn);
        (void)hipMemsetAsync(smax,  0, (size_t)Nn * HEADS * 4, stream);
        (void)hipMemsetAsync(denom, 0, (size_t)Nn * HEADS * 4, stream);
        (void)hipMemsetAsync(accb,  0, (size_t)Nn * FDIM * 4, stream);
        k_edge_score<<<gEdge, 256, 0, stream>>>(xl, xr, att + l * HEADS * CH,
                                                ei, E, Etot, score, smax);
        k_edge_exp<<<gEdge, 256, 0, stream>>>(ei, E, Etot, smax, score, denom);
        k_edge_aggr<<<gEdge, 256, 0, stream>>>(ei, E, Etot, xl, score, denom, accb);
        k_bias_relu<<<gElem, 256, 0, stream>>>(accb, bias + l * FDIM, hbuf, nElem);
        hin = hbuf;
    }
    k_head<<<(Nn + 7) / 8, 256, 0, stream>>>(hbuf, Wf, bf, out, Nn);
}
